// PlanarEdgeNetVAE_59339268162218
// MI455X (gfx1250) — matrix operand order repaired and checked by reading, not compiled, not run on hardware
//
#include <hip/hip_runtime.h>
#include <math.h>

// ---------------------------------------------------------------------------
// CDNA5 (gfx1250) implementation of PlanarEdgeNetVAE reference.
// wave32, WMMA f16 16x16x32 with f32 accumulation for all per-edge MLP GEMMs.
// BN affine folded into the next linear layer (weights+bias) or into the
// segment-mean epilogue, eliminating nearly all elementwise BN passes.
// ---------------------------------------------------------------------------

typedef _Float16 v16h __attribute__((ext_vector_type(16)));
typedef _Float16 v8h  __attribute__((ext_vector_type(8)));
typedef float    v8f  __attribute__((ext_vector_type(8)));

// ========================= utility kernels =================================

__global__ void zero_f32_kernel(float* __restrict__ p, long long n) {
    long long stride = (long long)blockDim.x * gridDim.x;
    for (long long i = (long long)blockIdx.x * blockDim.x + threadIdx.x; i < n; i += stride)
        p[i] = 0.f;
}

__global__ void edge_count_kernel(const int* __restrict__ dst, int E, float* __restrict__ cnt) {
    long long stride = (long long)blockDim.x * gridDim.x;
    for (long long e = (long long)blockIdx.x * blockDim.x + threadIdx.x; e < E; e += stride)
        atomicAdd(&cnt[dst[e]], 1.f);
}

// ========================= BatchNorm helpers ===============================

// Scalar stats (used for fp32 node input, C=4; C divides 256).
__global__ void bn_stats_f32_kernel(const float* __restrict__ buf, long long total, int C,
                                    float* __restrict__ sums /* [2*C] */) {
    __shared__ float ls[256], ls2[256];
    long long stride = (long long)blockDim.x * gridDim.x;
    float s = 0.f, s2 = 0.f;
    for (long long i = (long long)blockIdx.x * blockDim.x + threadIdx.x; i < total; i += stride) {
        float v = buf[i];
        s += v; s2 += v * v;
    }
    ls[threadIdx.x] = s; ls2[threadIdx.x] = s2;
    __syncthreads();
    int c = threadIdx.x;
    if (c < C) {
        float ts = 0.f, ts2 = 0.f;
        for (int j = c; j < 256; j += C) { ts += ls[j]; ts2 += ls2[j]; }
        atomicAdd(&sums[c], ts);
        atomicAdd(&sums[C + c], ts2);
    }
}

// Vectorized f16 stats: 8 elements (16B) per iteration, per-thread 8-channel
// accumulators. Requires C % 8 == 0, C <= 256, total % 8 == 0, blockDim 256.
__global__ void __launch_bounds__(256)
bn_stats_f16v_kernel(const _Float16* __restrict__ buf, long long total8, int C,
                     float* __restrict__ sums /* [2*C] */) {
    __shared__ float ls[256 * 8], ls2[256 * 8];
    float a[8] = {}, a2[8] = {};
    long long stride = (long long)blockDim.x * gridDim.x;
    for (long long i = (long long)blockIdx.x * blockDim.x + threadIdx.x; i < total8; i += stride) {
        v8h v = *(const v8h*)(buf + i * 8);
        #pragma unroll
        for (int j = 0; j < 8; j++) { float f = (float)v[j]; a[j] += f; a2[j] += f * f; }
    }
    #pragma unroll
    for (int j = 0; j < 8; j++) { ls[threadIdx.x * 8 + j] = a[j]; ls2[threadIdx.x * 8 + j] = a2[j]; }
    __syncthreads();
    int c = threadIdx.x;
    if (c < C) {
        // thread t slot j covers channel (t*8)%C + j
        int cb = c & ~7, j = c & 7, step = C >> 3;
        float ts = 0.f, ts2 = 0.f;
        for (int t = (cb >> 3); t < 256; t += step) { ts += ls[t * 8 + j]; ts2 += ls2[t * 8 + j]; }
        atomicAdd(&sums[c], ts);
        atomicAdd(&sums[C + c], ts2);
    }
}

__global__ void bn_finalize_kernel(const float* __restrict__ sums, int C, float invCnt,
                                   const float* __restrict__ g, const float* __restrict__ be,
                                   float* __restrict__ scale, float* __restrict__ shift) {
    int c = threadIdx.x;
    if (c < C) {
        float m = sums[c] * invCnt;
        float v = sums[C + c] * invCnt - m * m;   // biased variance (jnp.var ddof=0)
        float s = g[c] * rsqrtf(v + 1e-5f);
        scale[c] = s;
        shift[c] = be[c] - m * s;
    }
}

// Fold BN shift through the next linear layer: bfold[n] = b[n] + sum_k shift[k]*W[k,n]
__global__ void fold_bias_kernel(const float* __restrict__ W, const float* __restrict__ b,
                                 const float* __restrict__ shift, int K, int Nact,
                                 float* __restrict__ bfold) {
    int n = threadIdx.x;
    if (n < Nact) {
        float s = b[n];
        for (int k = 0; k < K; k++) s += shift[k] * W[k * Nact + n];
        bfold[n] = s;
    }
}

// Explicit BN apply (only needed for dec1 lin->BN->ReLU). C % 8 == 0.
__global__ void apply_bn_f16v_kernel(_Float16* __restrict__ buf, long long total8, int C,
                                     const float* __restrict__ scale, const float* __restrict__ shift,
                                     int relu) {
    long long stride = (long long)blockDim.x * gridDim.x;
    for (long long i = (long long)blockIdx.x * blockDim.x + threadIdx.x; i < total8; i += stride) {
        int c0 = (int)((i * 8) % C);
        v8h v = *(v8h*)(buf + i * 8);
        #pragma unroll
        for (int j = 0; j < 8; j++) {
            float f = (float)v[j] * scale[c0 + j] + shift[c0 + j];
            if (relu) f = fmaxf(f, 0.f);
            v[j] = (_Float16)f;
        }
        *(v8h*)(buf + i * 8) = v;
    }
}

__global__ void apply_bn_f32_kernel(const float* __restrict__ x, float* __restrict__ out,
                                    long long total, int C,
                                    const float* __restrict__ scale, const float* __restrict__ shift) {
    long long stride = (long long)blockDim.x * gridDim.x;
    for (long long i = (long long)blockIdx.x * blockDim.x + threadIdx.x; i < total; i += stride) {
        int c = (int)(i % C);
        out[i] = x[i] * scale[c] + shift[c];
    }
}

// ========================= aggregation (segment mean) ======================

__global__ void agg_edges_kernel(const _Float16* __restrict__ msg, const int* __restrict__ dst,
                                 int E, int C, float* __restrict__ agg) {
    long long stride = (long long)blockDim.x * gridDim.x;
    for (long long e = (long long)blockIdx.x * blockDim.x + threadIdx.x; e < E; e += stride) {
        int d = dst[e];
        const _Float16* mrow = msg + (size_t)e * C;
        float* arow = agg + (size_t)d * C;
        if ((C & 7) == 0) {
            for (int c0 = 0; c0 < C; c0 += 8) {
                v8h v = *(const v8h*)(mrow + c0);
                #pragma unroll
                for (int j = 0; j < 8; j++) atomicAdd(&arow[c0 + j], (float)v[j]);
            }
        } else {
            for (int c = 0; c < C; c++) atomicAdd(&arow[c], (float)mrow[c]);
        }
    }
}

// out = mean(msg) with optional folded BN affine: s*mean + (cnt>0 ? sh : 0).
__global__ void agg_div_kernel(const float* __restrict__ agg, const float* __restrict__ cnt,
                               int N, int C,
                               const float* __restrict__ scale, const float* __restrict__ shift,
                               float* __restrict__ out) {
    long long total = (long long)N * C;
    long long stride = (long long)blockDim.x * gridDim.x;
    for (long long i = (long long)blockIdx.x * blockDim.x + threadIdx.x; i < total; i += stride) {
        long long n = i / C;
        int c = (int)(i - n * C);
        float ct = cnt[n];
        float m = agg[i] / fmaxf(ct, 1.f);
        if (scale) m = m * scale[c] + ((ct > 0.f) ? shift[c] : 0.f);
        out[i] = m;
    }
}

// ========================= WMMA edge GEMMs =================================
// Flavor 1: EdgeConv first layer. Builds A = [h_i, h_j - h_i] (zero-padded to
// Kp) in LDS from node features + edge indices; C = relu?(A @ W + b).
// Block = 128 threads = 4 waves; each wave owns one 16-edge row tile.
template<int NP16>
__global__ void __launch_bounds__(128)
edge_gemm_gather(const float* __restrict__ nodeH, int D,
                 const int* __restrict__ src, const int* __restrict__ dst, int E,
                 const float* __restrict__ W, const float* __restrict__ bias,
                 int Kp, int Nact, _Float16* __restrict__ out, int relu)
{
    constexpr int Np = NP16 * 16;
    __shared__ alignas(32) _Float16 Wt[64 * 128];   // [n][k] transposed, zero padded
    __shared__ alignas(32) _Float16 As[64 * 128];   // [edge][k]
    __shared__ alignas(32) _Float16 Cs[64 * 64];    // [edge][n] output staging
    __shared__ int sdst[64], ssrc[64];
    const int tid  = threadIdx.x;
    const int lane = tid & 31;
    const int wv   = tid >> 5;
    const int rowl = lane & 15;
    const int hi   = lane >> 4;          // 0: lanes 0-15, 1: lanes 16-31
    const int Kact = 2 * D;

    for (int i = tid; i < Np * Kp; i += 128) {
        int n = i / Kp, k = i - n * Kp;
        Wt[i] = (_Float16)((k < Kact && n < Nact) ? W[k * Nact + n] : 0.f);
    }
    // zero the K padding region of As once (constant across tiles)
    for (int i = tid; i < (Kp << 6); i += 128) {
        int k = i % Kp;
        if (k >= Kact) As[i] = (_Float16)0.f;
    }

    const int tiles = (E + 63) >> 6;
    for (int tile = blockIdx.x; tile < tiles; tile += gridDim.x) {
        __syncthreads();
        const int e0 = tile << 6;
        if (tid < 64) {
            int e = e0 + tid;
            sdst[tid] = (e < E) ? dst[e] : 0;
            ssrc[tid] = (e < E) ? src[e] : 0;
        }
        __syncthreads();
        for (int i = tid; i < (D << 6); i += 128) {
            int el = i / D, k = i - el * D;
            float hd = nodeH[(size_t)sdst[el] * D + k];
            float hs = nodeH[(size_t)ssrc[el] * D + k];
            _Float16* arow = &As[el * Kp];
            arow[k]     = (_Float16)hd;
            arow[D + k] = (_Float16)(hs - hd);
        }
        __syncthreads();

        v8f acc[NP16] = {};
        const _Float16* arow = &As[(wv * 16 + rowl) * Kp];
        for (int kk = 0; kk < Kp; kk += 32) {
            // A frag: low lanes K kk+0..15, high lanes kk+16..31 (matches B)
            v16h a = *(const v16h*)(arow + kk + hi * 16);
            #pragma unroll
            for (int t = 0; t < NP16; t++) {
                v16h b = *(const v16h*)(&Wt[(t * 16 + rowl) * Kp + kk + hi * 16]);
                acc[t] = __builtin_amdgcn_wmma_f32_16x16x32_f16(
                    false, a, false, b, (short)0, acc[t], false, false);
            }
        }
        // scatter accumulators to LDS tile, then coalesced b128 stores
        #pragma unroll
        for (int t = 0; t < NP16; t++) {
            int col = t * 16 + rowl;
            if (col < Nact) {
                float bv = bias[col];
                #pragma unroll
                for (int r = 0; r < 8; r++) {
                    float v = acc[t][r] + bv;
                    if (relu) v = fmaxf(v, 0.f);
                    Cs[(wv * 16 + hi * 8 + r) * Nact + col] = (_Float16)v;
                }
            }
        }
        __syncthreads();
        const size_t obase = (size_t)e0 * Nact;
        if (e0 + 64 <= E) {
            for (int i = tid; i < (64 * Nact) / 8; i += 128)
                *(v8h*)(out + obase + (size_t)i * 8) = *(const v8h*)(Cs + i * 8);
        } else {
            for (int i = tid; i < 64 * Nact; i += 128)
                if (e0 + i / Nact < E) out[obase + i] = Cs[i];
        }
    }
}

// Flavor 2: A is a previous layer's f16 edge buffer [E, K] (K mult of 32).
// Optional per-input-channel scale (folded BN) applied while staging weights.
template<int NP16>
__global__ void __launch_bounds__(128)
edge_gemm(const _Float16* __restrict__ A, int K, int E,
          const float* __restrict__ W, const float* __restrict__ bias,
          const float* __restrict__ scaleK,
          int Nact, _Float16* __restrict__ out, int relu)
{
    constexpr int Np = NP16 * 16;
    __shared__ alignas(32) _Float16 Wt[64 * 64];
    __shared__ alignas(32) _Float16 Cs[64 * 64];
    const int tid  = threadIdx.x;
    const int lane = tid & 31;
    const int wv   = tid >> 5;
    const int rowl = lane & 15;
    const int hi   = lane >> 4;

    for (int i = tid; i < Np * K; i += 128) {
        int n = i / K, k = i - n * K;
        float w = (n < Nact) ? W[k * Nact + n] : 0.f;
        if (scaleK) w *= scaleK[k];
        Wt[i] = (_Float16)w;
    }
    __syncthreads();

    const int tiles = (E + 63) >> 6;
    for (int tile = blockIdx.x; tile < tiles; tile += gridDim.x) {
        const int e0 = tile << 6;
        const _Float16* arow = A + (size_t)(e0 + wv * 16 + rowl) * K;
        {   // prefetch next tile's A rows (global_prefetch_b8)
            int ntile = tile + gridDim.x;
            if (ntile < tiles)
                __builtin_prefetch(A + (size_t)((ntile << 6) + wv * 16 + rowl) * K, 0, 1);
        }
        v8f acc[NP16] = {};
        for (int kk = 0; kk < K; kk += 32) {
            v16h a = *(const v16h*)(arow + kk + hi * 16);
            #pragma unroll
            for (int t = 0; t < NP16; t++) {
                v16h b = *(const v16h*)(&Wt[(t * 16 + rowl) * K + kk + hi * 16]);
                acc[t] = __builtin_amdgcn_wmma_f32_16x16x32_f16(
                    false, a, false, b, (short)0, acc[t], false, false);
            }
        }
        #pragma unroll
        for (int t = 0; t < NP16; t++) {
            int col = t * 16 + rowl;
            if (col < Nact) {
                float bv = bias[col];
                #pragma unroll
                for (int r = 0; r < 8; r++) {
                    float v = acc[t][r] + bv;
                    if (relu) v = fmaxf(v, 0.f);
                    Cs[(wv * 16 + hi * 8 + r) * Nact + col] = (_Float16)v;
                }
            }
        }
        __syncthreads();
        const size_t obase = (size_t)e0 * Nact;
        if (e0 + 64 <= E) {
            for (int i = tid; i < (64 * Nact) / 8; i += 128)
                *(v8h*)(out + obase + (size_t)i * 8) = *(const v8h*)(Cs + i * 8);
        } else {
            for (int i = tid; i < 64 * Nact; i += 128)
                if (e0 + i / Nact < E) out[obase + i] = Cs[i];
        }
        __syncthreads();
    }
}

// ========================= per-node heads + planar flows ====================

__global__ void __launch_bounds__(256)
heads_flows_kernel(const float* __restrict__ nodeH, const float* __restrict__ eps,
                   const float* __restrict__ Wmu, const float* __restrict__ bmu,
                   const float* __restrict__ Wvar, const float* __restrict__ bvar,
                   const float* __restrict__ Wu,  const float* __restrict__ bu,
                   const float* __restrict__ Ww,  const float* __restrict__ bw,
                   const float* __restrict__ Wb,  const float* __restrict__ bb,
                   int N,
                   float* __restrict__ o_mu, float* __restrict__ o_lv, float* __restrict__ o_ldj,
                   float* __restrict__ o_z0, float* __restrict__ o_z, float* __restrict__ nodeZ)
{
    __shared__ float s[1122];
    const int oWmu = 0, oWvar = 64, oWu = 128, oWw = 512, oWb = 896;
    const int obmu = 1088, obvar = 1090, obu = 1092, obw = 1104, obb = 1116;
    for (int i = threadIdx.x; i < 1122; i += blockDim.x) {
        float v;
        if      (i < 64)   v = Wmu[i];
        else if (i < 128)  v = Wvar[i - 64];
        else if (i < 512)  v = Wu[i - 128];
        else if (i < 896)  v = Ww[i - 512];
        else if (i < 1088) v = Wb[i - 896];
        else if (i < 1090) v = bmu[i - 1088];
        else if (i < 1092) v = bvar[i - 1090];
        else if (i < 1104) v = bu[i - 1092];
        else if (i < 1116) v = bw[i - 1104];
        else               v = bb[i - 1116];
        s[i] = v;
    }
    __syncthreads();

    int n = blockIdx.x * blockDim.x + threadIdx.x;
    if (n >= N) return;

    float h[32];
    #pragma unroll
    for (int k = 0; k < 32; k++) h[k] = nodeH[(size_t)n * 32 + k];

    float mu0 = s[obmu], mu1 = s[obmu + 1], lv0 = s[obvar], lv1 = s[obvar + 1];
    float uu[12], ww[12], bbv[6];
    #pragma unroll
    for (int j = 0; j < 12; j++) { uu[j] = s[obu + j]; ww[j] = s[obw + j]; }
    #pragma unroll
    for (int j = 0; j < 6; j++)  bbv[j] = s[obb + j];

    #pragma unroll 4
    for (int k = 0; k < 32; k++) {
        float hk = h[k];
        mu0 += hk * s[oWmu + k * 2];     mu1 += hk * s[oWmu + k * 2 + 1];
        lv0 += hk * s[oWvar + k * 2];    lv1 += hk * s[oWvar + k * 2 + 1];
        #pragma unroll
        for (int j = 0; j < 12; j++) { uu[j] += hk * s[oWu + k * 12 + j]; ww[j] += hk * s[oWw + k * 12 + j]; }
        #pragma unroll
        for (int j = 0; j < 6; j++)  bbv[j] += hk * s[oWb + k * 6 + j];
    }

    float e0 = eps[(size_t)n * 2], e1 = eps[(size_t)n * 2 + 1];
    float z00 = mu0 + e0 * expf(0.5f * lv0);
    float z01 = mu1 + e1 * expf(0.5f * lv1);
    float zc0 = z00, zc1 = z01, ldj = 0.f;
    #pragma unroll
    for (int f = 0; f < 6; f++) {
        float u0 = uu[f * 2], u1 = uu[f * 2 + 1];
        float w0 = ww[f * 2], w1 = ww[f * 2 + 1];
        float bf = bbv[f];
        float uwv = w0 * u0 + w1 * u1;
        float sp  = (uwv > 20.f) ? uwv : log1pf(expf(uwv));   // softplus
        float c   = (-1.f + sp - uwv) / (w0 * w0 + w1 * w1);
        float uh0 = u0 + c * w0, uh1 = u1 + c * w1;
        float th  = tanhf(w0 * zc0 + w1 * zc1 + bf);
        zc0 += uh0 * th; zc1 += uh1 * th;
        float om = 1.f - th * th;
        ldj += logf(fabsf(1.f + (w0 * om) * uh0 + (w1 * om) * uh1));
    }
    o_mu[n * 2] = mu0;  o_mu[n * 2 + 1] = mu1;
    o_lv[n * 2] = lv0;  o_lv[n * 2 + 1] = lv1;
    o_ldj[n] = ldj;
    o_z0[n * 2] = z00;  o_z0[n * 2 + 1] = z01;
    o_z[n * 2]  = zc0;  o_z[n * 2 + 1]  = zc1;
    nodeZ[n * 2] = zc0; nodeZ[n * 2 + 1] = zc1;
}

// ========================= launch ==========================================

extern "C" void kernel_launch(void* const* d_in, const int* in_sizes, int n_in,
                              void* d_out, int out_size, void* d_ws, size_t ws_size,
                              hipStream_t stream)
{
    (void)n_in; (void)out_size; (void)ws_size;
    const int N = in_sizes[0] / 4;      // 50000 nodes
    const int E = in_sizes[1] / 2;      // 1000000 edges (multiple of 64)

    const float* x   = (const float*)d_in[0];
    const int*   ei  = (const int*)d_in[1];
    const float* eps = (const float*)d_in[2];
    const int* src = ei;
    const int* dst = ei + E;

    // params flattened in jax pytree order (dict keys sorted alphabetically)
    const float* amorb_b = (const float*)d_in[3];  const float* amorb_w = (const float*)d_in[4];
    const float* amoru_b = (const float*)d_in[5];  const float* amoru_w = (const float*)d_in[6];
    const float* amorw_b = (const float*)d_in[7];  const float* amorw_w = (const float*)d_in[8];
    const float* bn0_be  = (const float*)d_in[9];  const float* bn0_g   = (const float*)d_in[10];
    const float* d1bn2_be=(const float*)d_in[11];  const float* d1bn2_g =(const float*)d_in[12];
    const float* d1bn3_be=(const float*)d_in[13];  const float* d1bn3_g =(const float*)d_in[14];
    const float* d1l1_b  =(const float*)d_in[15];  const float* d1l1_w  =(const float*)d_in[16];
    const float* d1l2_b  =(const float*)d_in[17];  const float* d1l2_w  =(const float*)d_in[18];
    const float* d1l3_b  =(const float*)d_in[19];  const float* d1l3_w  =(const float*)d_in[20];
    const float* d2bn1_be=(const float*)d_in[21];  const float* d2bn1_g =(const float*)d_in[22];
    const float* d2bn2_be=(const float*)d_in[23];  const float* d2bn2_g =(const float*)d_in[24];
    const float* d2l1_b  =(const float*)d_in[25];  const float* d2l1_w  =(const float*)d_in[26];
    const float* d2l2_b  =(const float*)d_in[27];  const float* d2l2_w  =(const float*)d_in[28];
    const float* d2l3_b  =(const float*)d_in[29];  const float* d2l3_w  =(const float*)d_in[30];
    const float* e1bn1_be=(const float*)d_in[31];  const float* e1bn1_g =(const float*)d_in[32];
    const float* e1bn2_be=(const float*)d_in[33];  const float* e1bn2_g =(const float*)d_in[34];
    const float* e1bn3_be=(const float*)d_in[35];  const float* e1bn3_g =(const float*)d_in[36];
    const float* e1l1_b  =(const float*)d_in[37];  const float* e1l1_w  =(const float*)d_in[38];
    const float* e1l2_b  =(const float*)d_in[39];  const float* e1l2_w  =(const float*)d_in[40];
    const float* e1l3_b  =(const float*)d_in[41];  const float* e1l3_w  =(const float*)d_in[42];
    const float* e2bn1_be=(const float*)d_in[43];  const float* e2bn1_g =(const float*)d_in[44];
    const float* e2bn2_be=(const float*)d_in[45];  const float* e2bn2_g =(const float*)d_in[46];
    const float* e2l1_b  =(const float*)d_in[47];  const float* e2l1_w  =(const float*)d_in[48];
    const float* e2l2_b  =(const float*)d_in[49];  const float* e2l2_w  =(const float*)d_in[50];
    const float* mu_b    =(const float*)d_in[51];  const float* mu_w    =(const float*)d_in[52];
    const float* var_b   =(const float*)d_in[53];  const float* var_w   =(const float*)d_in[54];

    // workspace layout (~282 MB)
    char* wsb = (char*)d_ws;
    _Float16* EA = (_Float16*)wsb;                                   // E*64 f16
    _Float16* EB = (_Float16*)(wsb + (size_t)E * 64 * 2);            // E*64 f16
    float* nodeH   = (float*)(wsb + (size_t)E * 64 * 4);             // N*64 f32
    float* nodeAgg = nodeH   + (size_t)N * 64;                       // N*64 f32
    float* nodeZ   = nodeAgg + (size_t)N * 64;                       // N*2
    float* cnt     = nodeZ   + (size_t)N * 2;                        // N
    float* sums    = cnt + N;                                        // 128
    float* scale   = sums + 128;                                     // 64
    float* shift   = scale + 64;                                     // 64
    float* bfold   = shift + 64;                                     // 64

    float* o_xd  = (float*)d_out;
    float* o_mu  = o_xd  + (size_t)N * 4;
    float* o_lv  = o_mu  + (size_t)N * 2;
    float* o_ldj = o_lv  + (size_t)N * 2;
    float* o_z0  = o_ldj + N;
    float* o_z   = o_z0  + (size_t)N * 2;

    const int tiles = (E + 63) / 64;
    const int gG = tiles < 4096 ? tiles : 4096;

    // compute batch stats of buf[E,C] and fold with BN params into scale/shift
    auto bn_stats = [&](_Float16* buf, int C, const float* g, const float* be) {
        zero_f32_kernel<<<1, 256, 0, stream>>>(sums, 2 * C);
        bn_stats_f16v_kernel<<<1024, 256, 0, stream>>>(buf, (long long)E * C / 8, C, sums);
        bn_finalize_kernel<<<1, 64, 0, stream>>>(sums, C, 1.f / (float)E, g, be, scale, shift);
    };
    auto aggregate = [&](const _Float16* msg, int C, float* dsth,
                         const float* sc, const float* sh) {
        long long tot = (long long)N * C;
        zero_f32_kernel<<<(int)((tot + 255) / 256), 256, 0, stream>>>(nodeAgg, tot);
        agg_edges_kernel<<<4096, 256, 0, stream>>>(msg, dst, E, C, nodeAgg);
        agg_div_kernel<<<(int)((tot + 255) / 256), 256, 0, stream>>>(nodeAgg, cnt, N, C, sc, sh, dsth);
    };

    // --- in-degree counts (shared by all EdgeConvs) ---
    zero_f32_kernel<<<(N + 255) / 256, 256, 0, stream>>>(cnt, (long long)N);
    edge_count_kernel<<<4096, 256, 0, stream>>>(dst, E, cnt);

    // --- bn0 over nodes (tiny: applied explicitly in fp32) ---
    zero_f32_kernel<<<1, 256, 0, stream>>>(sums, 8);
    bn_stats_f32_kernel<<<512, 256, 0, stream>>>(x, (long long)N * 4, 4, sums);
    bn_finalize_kernel<<<1, 64, 0, stream>>>(sums, 4, 1.f / (float)N, bn0_g, bn0_be, scale, shift);
    apply_bn_f32_kernel<<<(int)(((long long)N * 4 + 255) / 256), 256, 0, stream>>>(
        x, nodeH, (long long)N * 4, 4, scale, shift);

    // --- EdgeConv 1 (enc1): 8 -> 64 -> 64 -> 32, relu then BN each layer ---
    edge_gemm_gather<4><<<gG, 128, 0, stream>>>(nodeH, 4, src, dst, E, e1l1_w, e1l1_b, 32, 64, EA, 1);
    bn_stats(EA, 64, e1bn1_g, e1bn1_be);                 // bn1 -> folded into l2
    fold_bias_kernel<<<1, 64, 0, stream>>>(e1l2_w, e1l2_b, shift, 64, 64, bfold);
    edge_gemm<4><<<gG, 128, 0, stream>>>(EA, 64, E, e1l2_w, bfold, scale, 64, EB, 1);
    bn_stats(EB, 64, e1bn2_g, e1bn2_be);                 // bn2 -> folded into l3
    fold_bias_kernel<<<1, 64, 0, stream>>>(e1l3_w, e1l3_b, shift, 64, 32, bfold);
    edge_gemm<2><<<gG, 128, 0, stream>>>(EB, 64, E, e1l3_w, bfold, scale, 32, EA, 1);
    bn_stats(EA, 32, e1bn3_g, e1bn3_be);                 // bn3 -> folded into mean
    aggregate(EA, 32, nodeH, scale, shift);

    // --- EdgeConv 2 (enc2): 64 -> 32 -> 32 ---
    edge_gemm_gather<2><<<gG, 128, 0, stream>>>(nodeH, 32, src, dst, E, e2l1_w, e2l1_b, 64, 32, EA, 1);
    bn_stats(EA, 32, e2bn1_g, e2bn1_be);                 // bn1 -> folded into l2
    fold_bias_kernel<<<1, 64, 0, stream>>>(e2l2_w, e2l2_b, shift, 32, 32, bfold);
    edge_gemm<2><<<gG, 128, 0, stream>>>(EA, 32, E, e2l2_w, bfold, scale, 32, EB, 1);
    bn_stats(EB, 32, e2bn2_g, e2bn2_be);                 // bn2 -> folded into mean
    aggregate(EB, 32, nodeH, scale, shift);

    // --- heads + reparameterization + 6 planar flows ---
    heads_flows_kernel<<<(N + 255) / 256, 256, 0, stream>>>(
        nodeH, eps, mu_w, mu_b, var_w, var_b,
        amoru_w, amoru_b, amorw_w, amorw_b, amorb_w, amorb_b, N,
        o_mu, o_lv, o_ldj, o_z0, o_z, nodeZ);

    // --- EdgeConv 3 (dec1): 4 -> 32(relu) -> 32(BN,relu) -> 64(relu,BN) ---
    edge_gemm_gather<2><<<gG, 128, 0, stream>>>(nodeZ, 2, src, dst, E, d1l1_w, d1l1_b, 32, 32, EA, 1);
    edge_gemm<2><<<gG, 128, 0, stream>>>(EA, 32, E, d1l2_w, d1l2_b, nullptr, 32, EB, 0); // pre-act
    bn_stats(EB, 32, d1bn2_g, d1bn2_be);
    apply_bn_f16v_kernel<<<2048, 256, 0, stream>>>(EB, (long long)E * 32 / 8, 32, scale, shift, 1);
    edge_gemm<4><<<gG, 128, 0, stream>>>(EB, 32, E, d1l3_w, d1l3_b, nullptr, 64, EA, 1);
    bn_stats(EA, 64, d1bn3_g, d1bn3_be);                 // bn3 -> folded into mean
    aggregate(EA, 64, nodeH, scale, shift);

    // --- EdgeConv 4 (dec2): 128 -> 64 -> 64 -> 4 ---
    edge_gemm_gather<4><<<gG, 128, 0, stream>>>(nodeH, 64, src, dst, E, d2l1_w, d2l1_b, 128, 64, EA, 1);
    bn_stats(EA, 64, d2bn1_g, d2bn1_be);                 // bn1 -> folded into l2
    fold_bias_kernel<<<1, 64, 0, stream>>>(d2l2_w, d2l2_b, shift, 64, 64, bfold);
    edge_gemm<4><<<gG, 128, 0, stream>>>(EA, 64, E, d2l2_w, bfold, scale, 64, EB, 1);
    bn_stats(EB, 64, d2bn2_g, d2bn2_be);                 // bn2 -> folded into l3
    fold_bias_kernel<<<1, 64, 0, stream>>>(d2l3_w, d2l3_b, shift, 64, 4, bfold);
    edge_gemm<1><<<gG, 128, 0, stream>>>(EB, 64, E, d2l3_w, bfold, scale, 4, EA, 0);
    aggregate(EA, 4, o_xd, nullptr, nullptr);
}
